// AR_48395691491547
// MI455X (gfx1250) — matrix operand order repaired and checked by reading, not compiled, not run on hardware
//
#include <hip/hip_runtime.h>

typedef __attribute__((ext_vector_type(16))) _Float16 v16h;
typedef __attribute__((ext_vector_type(8)))  _Float16 v8h;
typedef __attribute__((ext_vector_type(8)))  float    v8f;

#define B_   64
#define T_   8192
#define D_   32      // hidden
#define G_   128     // 4*D gates
#define C_   256     // output categories
#define CHUNK 128    // x staging chunk (steps)

// ---- fast activations: prefer native v_tanh_f32 (gfx1250), else exp + v_rcp_f32
static __device__ __forceinline__ float tanh_approx(float x) {
#if __has_builtin(__builtin_amdgcn_tanhf)
    return __builtin_amdgcn_tanhf(x);
#elif __has_builtin(__builtin_amdgcn_tanh_f32)
    return __builtin_amdgcn_tanh_f32(x);
#else
    x = fminf(fmaxf(x, -15.0f), 15.0f);
    float e = __expf(2.0f * x);
    return (e - 1.0f) * __builtin_amdgcn_rcpf(e + 1.0f);
#endif
}
static __device__ __forceinline__ float sigm(float x) {
#if __has_builtin(__builtin_amdgcn_tanhf) || __has_builtin(__builtin_amdgcn_tanh_f32)
    return fmaf(0.5f, tanh_approx(0.5f * x), 0.5f);
#else
    return __builtin_amdgcn_rcpf(1.0f + __expf(-x));
#endif
}

// Build the 16x32 f16 A-operand (WMMA layout) from a row-major [16][32] f16 block.
// lanes 0-15: row=lane,  elems 0..15 = K0..15
// lanes 16-31: row=lane-16, elems 0..15 = K16..31
static __device__ __forceinline__ v16h load_a16(const _Float16* base, int lane) {
    const bool lo = (lane & 16) == 0;
    const int  r  = lane & 15;
    const _Float16* p = base + r * D_;
    v8h a0 = *(const v8h*)(p + (lo ? 0 : 16));
    v8h a1 = *(const v8h*)(p + (lo ? 8 : 24));
    v16h a;
#pragma unroll
    for (int j = 0; j < 8; ++j) { a[j] = a0[j]; a[j + 8] = a1[j]; }
    return a;
}

// ---------------------------------------------------------------------------
// Kernel 1: LSTM scan. 4 blocks x 256 threads; block handles 16 batch rows.
// Per step: gates = h_{t-1} @ Wh  (WMMA, one 16x16 tile per wave, K=32)
//           + (s * v + u) injected via the WMMA C operand.
// ---------------------------------------------------------------------------
__global__ __launch_bounds__(256) void lstm_scan_kernel(
    const int*   __restrict__ x,       // [B,T]
    const float* __restrict__ bos,     // [D]
    const float* __restrict__ W_in,    // [1,D]
    const float* __restrict__ b_in,    // [D]
    const float* __restrict__ Wi,      // [D,4D]
    const float* __restrict__ Wh,      // [D,4D]
    const float* __restrict__ b_lstm,  // [4D]
    _Float16*    __restrict__ hs)      // [B,T,D] f16 (workspace)
{
    __shared__ __align__(64) _Float16 hA[16 * D_];     // h state, WMMA-A source
    __shared__ float gbuf[16 * G_];                    // gate exchange
    __shared__ float sbuf[16 * CHUNK];                 // staged input scalars

    const int tid  = threadIdx.x;
    const int lane = tid & 31;
    const int wv   = tid >> 5;              // wave 0..7 -> gate columns [16w,16w+16)
    const int b0   = blockIdx.x * 16;

    // ---- prologue: per-lane fused input vectors -----------------------------
    const int ncol = wv * 16 + (lane & 15); // global gate column for this lane's N
    float v_n = 0.f, u_n = 0.f, g0_n = 0.f;
    for (int d = 0; d < D_; ++d) {
        float wi = Wi[d * G_ + ncol];
        v_n  += W_in[d] * wi;
        u_n  += b_in[d] * wi;
        g0_n += bos[d]  * wi;
    }
    u_n  += b_lstm[ncol];
    g0_n += b_lstm[ncol];

    // ---- B operand: Wh column tile, kept in registers for all 8192 steps ----
    v16h bmat;
    {
        const int kbase = (lane < 16) ? 0 : 16;
#pragma unroll
        for (int j = 0; j < 16; ++j)
            bmat[j] = (_Float16)Wh[(kbase + j) * G_ + ncol];
    }

    // ---- per-thread cell state: elements (em, ed), (em, ed+1) ---------------
    const int em = tid >> 4;          // batch row within tile, 0..15
    const int ed = (tid & 15) * 2;    // even hidden index, 0..30
    const int mbase = (lane < 16) ? 0 : 8;
    float c0 = 0.f, c1 = 0.f;

    // zero h state (256 dwords == 16x32 halves)
    ((unsigned*)hA)[tid] = 0u;

    // stage s[m][j] = xf(x[b0+m][t0+j-1]) for chunk starting at t0
    auto stage_chunk = [&](int t0) {
        __syncthreads();
#pragma unroll
        for (int k = 0; k < (16 * CHUNK) / 256; ++k) {
            int e = tid + k * 256;
            int m = e >> 7;               // e / CHUNK
            int j = e & (CHUNK - 1);
            int ti = t0 + j - 1;
            float s = 0.f;
            if (ti >= 0)
                s = (float)x[(b0 + m) * T_ + ti] * (1.0f / 255.0f) - 0.5f;
            sbuf[m * CHUNK + j] = s;
            if (t0 + CHUNK < T_)   // warm next chunk's lines
                __builtin_prefetch((const void*)&x[(b0 + m) * T_ + ti + CHUNK], 0, 0);
        }
        __syncthreads();
    };

    // one recurrence step: WMMA + gate exchange + elementwise update
    auto do_step = [&](const v8f& cop, int t) {
        // gates(16x16 tile) = h_{t-1}(16x32) @ Wh_tile(32x16) + cop
        v16h amat = load_a16(hA, lane);
        v8f  dres = __builtin_amdgcn_wmma_f32_16x16x32_f16(
                        false, amat, false, bmat, (short)0, cop, false, false);
#pragma unroll
        for (int r = 0; r < 8; ++r)
            gbuf[(mbase + r) * G_ + ncol] = dres[r];
        __syncthreads();

        // elementwise LSTM update (gate order i,f,g,o across columns)
        const float* gm = &gbuf[em * G_];
        float xi0 = gm[ed],            xi1 = gm[ed + 1];
        float xf0 = gm[D_ + ed],       xf1 = gm[D_ + ed + 1];
        float xg0 = gm[2 * D_ + ed],   xg1 = gm[2 * D_ + ed + 1];
        float xo0 = gm[3 * D_ + ed],   xo1 = gm[3 * D_ + ed + 1];
        c0 = sigm(xf0) * c0 + sigm(xi0) * tanh_approx(xg0);
        c1 = sigm(xf1) * c1 + sigm(xi1) * tanh_approx(xg1);
        float h0 = sigm(xo0) * tanh_approx(c0);
        float h1 = sigm(xo1) * tanh_approx(c1);
        union { _Float16 h[2]; unsigned u; } pk;
        pk.h[0] = (_Float16)h0;
        pk.h[1] = (_Float16)h1;
        *(unsigned*)&hA[em * D_ + ed] = pk.u;
        *(unsigned*)(hs + (((b0 + em) * T_ + t) * D_ + ed)) = pk.u;
        __syncthreads();
    };

    // ---- step 0 (bos path), peeled -----------------------------------------
    stage_chunk(0);                        // also covers hA zero-init sync
    {
        v8f cop;
#pragma unroll
        for (int r = 0; r < 8; ++r) cop[r] = g0_n;
        do_step(cop, 0);
    }

    // ---- steps 1..T-1 -------------------------------------------------------
    for (int t = 1; t < T_; ++t) {
        if ((t & (CHUNK - 1)) == 0)
            stage_chunk(t);

        const int jj = t & (CHUNK - 1);
        v8f cop;
#pragma unroll
        for (int r = 0; r < 8; ++r) {
            float s = sbuf[(mbase + r) * CHUNK + jj];
            cop[r] = fmaf(s, v_n, u_n);
        }
        do_step(cop, t);
    }
}

// ---------------------------------------------------------------------------
// Kernel 2: output projection  out[B*T, 256] = hs[B*T, 32] @ W_out + b_out.
// Bandwidth bound (536 MB of f32 out). One WMMA (K=32) per 16x16 tile,
// bias via the C operand, B operand hoisted to registers, NT stores.
// ---------------------------------------------------------------------------
__global__ __launch_bounds__(256) void out_gemm_kernel(
    const _Float16* __restrict__ hs,     // [B*T, D]
    const float*    __restrict__ W_out,  // [D, C]
    const float*    __restrict__ b_out,  // [C]
    float*          __restrict__ out)    // [B*T, C]
{
    const int tid  = threadIdx.x;
    const int lane = tid & 31;
    const int wv   = tid >> 5;
    const long long rowBlock = (long long)blockIdx.x * 256;

    const int n     = lane & 15;
    const int kbase = (lane < 16) ? 0 : 16;
    const int colA  = wv * 32 + n;        // wave covers N-tiles 2w and 2w+1
    const int colB  = colA + 16;

    v16h bm0, bm1;
#pragma unroll
    for (int j = 0; j < 16; ++j) {
        bm0[j] = (_Float16)W_out[(kbase + j) * C_ + colA];
        bm1[j] = (_Float16)W_out[(kbase + j) * C_ + colB];
    }
    const float bias0 = b_out[colA];
    const float bias1 = b_out[colB];
    v8f cop0, cop1;
#pragma unroll
    for (int r = 0; r < 8; ++r) { cop0[r] = bias0; cop1[r] = bias1; }

    const int mbase = (lane < 16) ? 0 : 8;

    for (int mt = 0; mt < 16; ++mt) {
        const long long r0 = rowBlock + (long long)mt * 16;
        const _Float16* ap = hs + r0 * D_;
        __builtin_prefetch((const void*)(ap + 16 * D_), 0, 0);

        v16h amat = load_a16(ap, lane);
        v8f d0 = __builtin_amdgcn_wmma_f32_16x16x32_f16(
                     false, amat, false, bm0, (short)0, cop0, false, false);
        v8f d1 = __builtin_amdgcn_wmma_f32_16x16x32_f16(
                     false, amat, false, bm1, (short)0, cop1, false, false);

#pragma unroll
        for (int r = 0; r < 8; ++r) {
            long long row = r0 + mbase + r;
            __builtin_nontemporal_store(d0[r], &out[row * C_ + colA]);
            __builtin_nontemporal_store(d1[r], &out[row * C_ + colB]);
        }
    }
}

extern "C" void kernel_launch(void* const* d_in, const int* in_sizes, int n_in,
                              void* d_out, int out_size, void* d_ws, size_t ws_size,
                              hipStream_t stream) {
    const int*   x      = (const int*)  d_in[0];
    const float* bos    = (const float*)d_in[1];
    const float* W_in   = (const float*)d_in[2];
    const float* b_in   = (const float*)d_in[3];
    const float* Wi     = (const float*)d_in[4];
    const float* Wh     = (const float*)d_in[5];
    const float* b_lstm = (const float*)d_in[6];
    const float* W_out  = (const float*)d_in[7];
    const float* b_out  = (const float*)d_in[8];
    float*     out = (float*)d_out;
    _Float16*  hs  = (_Float16*)d_ws;   // [B,T,D] f16 = 32 MB scratch

    lstm_scan_kernel<<<B_ / 16, 256, 0, stream>>>(
        x, bos, W_in, b_in, Wi, Wh, b_lstm, hs);

    out_gemm_kernel<<<(B_ * T_) / 256, 256, 0, stream>>>(
        hs, W_out, b_out, out);
}
